// RetrieverSupervised_20658792693954
// MI455X (gfx1250) — matrix operand order repaired and checked by reading, not compiled, not run on hardware
//
#include <hip/hip_runtime.h>
#include <stdint.h>

#define N_Q   32
#define N_DOCS 8
#define N_M   (N_Q * N_DOCS)   // 256 docs
#define Q_LEN 32
#define D_LEN 200
#define D_PAD 208              // 13 * 16, padded for clean 16-row WMMA tiles
#define DIM   128
#define NTILE_D (D_PAD / 16)   // 13
#define NG    8                // queries handled per block

typedef __bf16 bf16_t;
typedef __attribute__((ext_vector_type(16))) bf16_t v16bf;
typedef __attribute__((ext_vector_type(8)))  float  v8f;

union Frag { v16bf v; uint4 u[2]; };

__device__ __forceinline__ unsigned short f32_to_bf16_rne(float f) {
  unsigned int x = __float_as_uint(f);
  unsigned int r = x + 0x7FFFu + ((x >> 16) & 1u);
  return (unsigned short)(r >> 16);
}

__device__ __forceinline__ float max8(const v8f& a) {
  float m01 = fmaxf(a[0], a[1]);
  float m23 = fmaxf(a[2], a[3]);
  float m45 = fmaxf(a[4], a[5]);
  float m67 = fmaxf(a[6], a[7]);
  return fmaxf(fmaxf(m01, m23), fmaxf(m45, m67));
}

// ---- pass 1a: convert hq fp32 -> bf16 (x4 vectorized) --------------------
__global__ void convert_hq(const float* __restrict__ hq,
                           unsigned short* __restrict__ dst) {
  int t = (blockIdx.x * blockDim.x + threadIdx.x) * 4;
  if (t < N_Q * Q_LEN * DIM) {
    float4 v = *(const float4*)(hq + t);
    ushort4 o;
    o.x = f32_to_bf16_rne(v.x); o.y = f32_to_bf16_rne(v.y);
    o.z = f32_to_bf16_rne(v.z); o.w = f32_to_bf16_rne(v.w);
    *(ushort4*)(dst + t) = o;
  }
}

// ---- pass 1b: convert hd fp32 -> bf16, pad 200 -> 208 rows (x4) ----------
__global__ void convert_hd_pad(const float* __restrict__ hd,
                               unsigned short* __restrict__ dst) {
  int t = (blockIdx.x * blockDim.x + threadIdx.x) * 4;  // padded-layout index
  if (t >= N_M * D_PAD * DIM) return;
  int h = t % DIM;                  // 4-aligned, same row for all 4 elems
  int r = (t / DIM) % D_PAD;
  int m = t / (DIM * D_PAD);
  float4 v = make_float4(0.f, 0.f, 0.f, 0.f);
  if (r < D_LEN) v = *(const float4*)(hd + ((size_t)m * D_LEN + r) * DIM + h);
  ushort4 o;
  o.x = f32_to_bf16_rne(v.x); o.y = f32_to_bf16_rne(v.y);
  o.z = f32_to_bf16_rne(v.z); o.w = f32_to_bf16_rne(v.w);
  *(ushort4*)(dst + t) = o;
}

// ---- pass 2: maxsim scores via bf16 WMMA ---------------------------------
// grid = (N_M, N_Q/NG); block = 256 threads (8 waves).
// Each block: one doc m vs NG=8 queries. Waves own 1-2 doc-token M-tiles
// resident in registers across the whole query loop; running maxes for all
// 8 queries stay in registers; a single LDS reduction finalizes the block.
__global__ __launch_bounds__(256)
void colbert_maxsim(const unsigned short* __restrict__ hqb,   // [N_Q][Q_LEN][DIM] bf16
                    const unsigned short* __restrict__ hdb,   // [N_M][D_PAD][DIM] bf16
                    float* __restrict__ score)                // [N_Q][N_M]
{
  const int m    = blockIdx.x;        // doc
  const int g    = blockIdx.y;        // query group
  const int tid  = threadIdx.x;
  // scalarize the wave id so tile-ownership branches are s_cbranch, not exec
  const int w    = __builtin_amdgcn_readfirstlane(tid >> 5);   // wave 0..7
  const int lane = tid & 31;
  const int lr   = lane & 15;
  const int hi   = lane >> 4;         // lane half
  const int kb   = hi * 16;           // A-matrix K base for this lane half
  const int rowoff = hi * 8;          // D-tile row offset for this lane half

  // Tile ownership: waves 0-4 own tiles {w, w+8}; waves 5-7 own tile {w}.
  const bool two_tiles = (w < 5);     // scalar condition

  // ---- preload doc (A) fragments once per block -------------------------
  Frag ad[2][4];
  const unsigned short* dptr = hdb + (size_t)m * D_PAD * DIM;
  {
    const unsigned short* arow = dptr + (size_t)(w * 16 + lr) * DIM;
    #pragma unroll
    for (int c = 0; c < 4; ++c) {
      ad[0][c].u[0] = *(const uint4*)(arow + 32 * c + kb);
      ad[0][c].u[1] = *(const uint4*)(arow + 32 * c + 8 + kb);
    }
  }
  if (two_tiles) {
    const unsigned short* arow = dptr + (size_t)((w + 8) * 16 + lr) * DIM;
    #pragma unroll
    for (int c = 0; c < 4; ++c) {
      ad[1][c].u[0] = *(const uint4*)(arow + 32 * c + kb);
      ad[1][c].u[1] = *(const uint4*)(arow + 32 * c + 8 + kb);
    }
  }
  // Tail: only tile 12 (wave 4, second tile) has invalid rows, exactly for
  // lanes 16-31 (rows 200-207) -> uniform per lane half.
  const bool valid1 = ((w + 8) * 16 + rowoff) < D_LEN;

  // ---- query loop: pure load -> WMMA -> max3, no barriers ---------------
  float rm[NG][2];
  #pragma unroll
  for (int i = 0; i < NG; ++i) { rm[i][0] = -__builtin_inff(); rm[i][1] = -__builtin_inff(); }

  #pragma unroll
  for (int i = 0; i < NG; ++i) {
    const int n = g * NG + i;

    // B 32x16 bf16 layout: lane = query column, 16 contiguous K at 16*hi.
    Frag bq[2][4];
    const unsigned short* qbase = hqb + ((size_t)n * Q_LEN + lr) * DIM + hi * 16;
    #pragma unroll
    for (int qt = 0; qt < 2; ++qt) {
      const unsigned short* qrow = qbase + qt * 16 * DIM;
      #pragma unroll
      for (int c = 0; c < 4; ++c) {
        const uint4* pp = (const uint4*)(qrow + 32 * c);
        bq[qt][c].u[0] = pp[0];
        bq[qt][c].u[1] = pp[1];
      }
    }

    {   // tile 0: always fully valid
      v8f acc0 = {}, acc1 = {};
      #pragma unroll
      for (int c = 0; c < 4; ++c) {
        acc0 = __builtin_amdgcn_wmma_f32_16x16x32_bf16(
            false, ad[0][c].v, false, bq[0][c].v, (short)0, acc0, false, false);
        acc1 = __builtin_amdgcn_wmma_f32_16x16x32_bf16(
            false, ad[0][c].v, false, bq[1][c].v, (short)0, acc1, false, false);
      }
      rm[i][0] = fmaxf(rm[i][0], max8(acc0));
      rm[i][1] = fmaxf(rm[i][1], max8(acc1));
    }
    if (two_tiles) {   // scalar branch; tile 12 masked per lane half
      v8f acc0 = {}, acc1 = {};
      #pragma unroll
      for (int c = 0; c < 4; ++c) {
        acc0 = __builtin_amdgcn_wmma_f32_16x16x32_bf16(
            false, ad[1][c].v, false, bq[0][c].v, (short)0, acc0, false, false);
        acc1 = __builtin_amdgcn_wmma_f32_16x16x32_bf16(
            false, ad[1][c].v, false, bq[1][c].v, (short)0, acc1, false, false);
      }
      float t0 = max8(acc0);
      float t1 = max8(acc1);
      rm[i][0] = valid1 ? fmaxf(rm[i][0], t0) : rm[i][0];
      rm[i][1] = valid1 ? fmaxf(rm[i][1], t1) : rm[i][1];
    }
  }

  // ---- single finalize: max over waves & lane halves, mean over q -------
  __shared__ float smax[8][NG][2][32];   // 16 KB
  #pragma unroll
  for (int i = 0; i < NG; ++i) {
    smax[w][i][0][lane] = rm[i][0];
    smax[w][i][1][lane] = rm[i][1];
  }
  __syncthreads();
  {
    // wave w finalizes query set i = w; one lane per query token
    const int qt = lane >> 4, c = lane & 15;
    float v = -__builtin_inff();
    #pragma unroll
    for (int ww = 0; ww < 8; ++ww)
      v = fmaxf(v, fmaxf(smax[ww][w][qt][c], smax[ww][w][qt][c + 16]));
    #pragma unroll
    for (int off = 16; off > 0; off >>= 1)
      v += __shfl_xor(v, off, 32);
    if (lane == 0)
      score[(size_t)(g * NG + w) * N_M + m] = v * (1.0f / Q_LEN);
  }
}

// ---- pass 3: row log-softmax + NLL loss (single block, no atomics) -------
__global__ __launch_bounds__(256)
void softmax_loss(float* __restrict__ out) {
  __shared__ float red[256];
  const int tid = threadIdx.x;
  float loss = 0.f;
  for (int nrow = 0; nrow < N_Q; ++nrow) {
    float s = out[nrow * N_M + tid];
    red[tid] = s; __syncthreads();
    for (int st = 128; st > 0; st >>= 1) {
      if (tid < st) red[tid] = fmaxf(red[tid], red[tid + st]);
      __syncthreads();
    }
    float mx = red[0]; __syncthreads();
    red[tid] = __expf(s - mx); __syncthreads();
    for (int st = 128; st > 0; st >>= 1) {
      if (tid < st) red[tid] += red[tid + st];
      __syncthreads();
    }
    if (tid == 0) {
      float st = out[nrow * N_M + nrow * N_DOCS];  // positive doc column
      loss += -(st - mx - __logf(red[0]));
    }
    __syncthreads();
  }
  if (tid == 0) out[N_Q * N_M] = loss * (1.0f / N_Q);
}

extern "C" void kernel_launch(void* const* d_in, const int* in_sizes, int n_in,
                              void* d_out, int out_size, void* d_ws, size_t ws_size,
                              hipStream_t stream) {
  const float* hq = (const float*)d_in[0];   // [32,32,128]  f32
  const float* hd = (const float*)d_in[1];   // [256,200,128] f32
  float* out = (float*)d_out;                // 8192 scores + 1 loss

  unsigned short* ws_hq = (unsigned short*)d_ws;                 // 256 KB
  unsigned short* ws_hd = ws_hq + (size_t)N_Q * Q_LEN * DIM;     // ~13.3 MB

  const int hq_elems = N_Q * Q_LEN * DIM;      // 131072
  const int hd_elems = N_M * D_PAD * DIM;      // 6815744 (padded)

  convert_hq<<<(hq_elems / 4 + 255) / 256, 256, 0, stream>>>(hq, ws_hq);
  convert_hd_pad<<<(hd_elems / 4 + 255) / 256, 256, 0, stream>>>(hd, ws_hd);

  dim3 grid(N_M, N_Q / NG);                    // 256 x 4 blocks
  colbert_maxsim<<<grid, 256, 0, stream>>>(ws_hq, ws_hd, out);
  softmax_loss<<<1, 256, 0, stream>>>(out);
  (void)in_sizes; (void)n_in; (void)out_size; (void)ws_size;
}
